// OrderedUnmerger_38551626448918
// MI455X (gfx1250) — matrix operand order repaired and checked by reading, not compiled, not run on hardware
//
#include <hip/hip_runtime.h>
#include <math.h>

typedef __attribute__((ext_vector_type(16))) __bf16 v16bf;
typedef __attribute__((ext_vector_type(8)))  __bf16 v8bf;
typedef __attribute__((ext_vector_type(8)))  float  v8f;

namespace {
constexpr int Ec = 256, Hh = 8, Fc = 1024, LEc = 4, LDc = 3, Kq = 8, CDc = 4;
constexpr int Bc = 16, Sc = 512;
constexpr int Mrows = Bc * Sc;                    // 8192
constexpr float LN_EPS   = 1e-5f;
constexpr float SIG2v    = 1.0f + 1e-8f;
constexpr float HD_SCALE = 0.17677669529663688f;  // 1/sqrt(32)
constexpr float BN_SCALE = 0.9999950000374997f;   // 1/sqrt(1+1e-5)
}

__device__ __forceinline__ float gelu_f(float x) {
  return 0.5f * x * (1.0f + erff(x * 0.7071067811865476f));
}
__device__ __forceinline__ float wave_max(float v) {
#pragma unroll
  for (int off = 16; off > 0; off >>= 1) v = fmaxf(v, __shfl_xor(v, off, 32));
  return v;
}
__device__ __forceinline__ float wave_sum(float v) {
#pragma unroll
  for (int off = 16; off > 0; off >>= 1) v += __shfl_xor(v, off, 32);
  return v;
}
__device__ __forceinline__ float nan0(float v) {
  if (isnan(v)) return 0.f;
  return fminf(fmaxf(v, -3.402823466e38f), 3.402823466e38f);
}

// ---------------- fp32 -> bf16 conversion (weights once per launch; memory once) ----------------
__global__ void cvt_bf16_k(const float* __restrict__ in, __bf16* __restrict__ out, int n) {
  int idx = blockIdx.x * blockDim.x + threadIdx.x;
  if (idx < n) out[idx] = (__bf16)in[idx];
}

// ---------------- input projection + BN(eval) + GELU + coords ----------------
__global__ void input_proj_k(const float* __restrict__ x,
                             const float* __restrict__ in_w, const float* __restrict__ in_b,
                             const float* __restrict__ bn_w, const float* __restrict__ bn_b,
                             float* __restrict__ h, float* __restrict__ coords) {
  int idx = blockIdx.x * blockDim.x + threadIdx.x;
  if (idx >= Mrows * Ec) return;
  int row = idx >> 8, e = idx & 255;
  const float* xr = x + (size_t)row * 4;
  float x0 = nan0(xr[0]), x1 = nan0(xr[1]), x2 = nan0(xr[2]), x3 = nan0(xr[3]);
  float a = in_b[e] + x0 * in_w[e * 4 + 0] + x1 * in_w[e * 4 + 1] +
            x2 * in_w[e * 4 + 2] + x3 * in_w[e * 4 + 3];
  a = a * BN_SCALE * bn_w[e] + bn_b[e];
  h[idx] = gelu_f(a);
  if (e == 0) { coords[2 * row] = x1; coords[2 * row + 1] = x2; }
}

// ---------------- LayerNorm: fp32 in, bf16 out (feeds WMMA GEMM A operand) ----------------
__global__ void layernorm_k(const float* __restrict__ in, __bf16* __restrict__ out,
                            const float* __restrict__ w, const float* __restrict__ b, int M) {
  int wave = threadIdx.x >> 5, lane = threadIdx.x & 31;
  int row = blockIdx.x * 8 + wave;
  if (row >= M) return;
  const float* r = in + (size_t)row * Ec;
  float vals[8], s = 0.f;
#pragma unroll
  for (int i = 0; i < 8; ++i) { vals[i] = r[lane + i * 32]; s += vals[i]; }
  float mean = wave_sum(s) * (1.0f / Ec);
  float v = 0.f;
#pragma unroll
  for (int i = 0; i < 8; ++i) { float d = vals[i] - mean; v += d * d; }
  float rstd = rsqrtf(wave_sum(v) * (1.0f / Ec) + LN_EPS);
  __bf16* o = out + (size_t)row * Ec;
#pragma unroll
  for (int i = 0; i < 8; ++i) {
    int e = lane + i * 32;
    o[e] = (__bf16)((vals[i] - mean) * rstd * w[e] + b[e]);
  }
}

// ---------------- WMMA GEMM: C[M,N] = op(A[M,K] @ W[N,K]^T + bias) (+res) ----------------
// A, W are bf16 (pre-converted); C is fp32 (residual path) or bf16 (feeds next GEMM).
template <bool GELU, bool RES, typename OutT>
__global__ __launch_bounds__(256) void gemm_k(const __bf16* __restrict__ A,
                                              const __bf16* __restrict__ W,
                                              const float* __restrict__ bias,
                                              const float* __restrict__ res,
                                              OutT* __restrict__ C,
                                              int M, int N, int Kd) {
  int lane = threadIdx.x & 31, wave = threadIdx.x >> 5;
  int wm = wave & 3, wn = wave >> 2;            // 4x2 waves => 64x64 tile
  int m0 = blockIdx.y * 64 + wm * 16;
  int n0 = blockIdx.x * 64 + wn * 32;
  int l16 = lane & 15, half = lane >> 4;
  int arow = m0 + l16; if (arow >= M) arow = M - 1;
  const __bf16* Ar  = A + (size_t)arow * Kd;
  const __bf16* Wn0 = W + (size_t)(n0 + l16) * Kd;
  const __bf16* Wn1 = W + (size_t)(n0 + 16 + l16) * Kd;
  v8f c0 = {}, c1 = {};
  for (int kk = 0; kk < Kd; kk += 32) {
    int ka = kk + half * 16;
    v8bf alo = *(const v8bf*)(Ar + ka);
    v8bf ahi = *(const v8bf*)(Ar + ka + 8);
    v16bf a = __builtin_shufflevector(alo, ahi, 0, 1, 2, 3, 4, 5, 6, 7,
                                      8, 9, 10, 11, 12, 13, 14, 15);
    int kb = kk + half * 16;
    v16bf b0 = *(const v16bf*)(Wn0 + kb);
    v16bf b1 = *(const v16bf*)(Wn1 + kb);
    if (kk + 32 < Kd) {   // prefetch next K-slice
      __builtin_prefetch(Ar + ka + 32, 0, 0);
      __builtin_prefetch(Wn0 + kb + 32, 0, 0);
      __builtin_prefetch(Wn1 + kb + 32, 0, 0);
    }
    c0 = __builtin_amdgcn_wmma_f32_16x16x32_bf16(false, a, false, b0, (short)0, c0, false, false);
    c1 = __builtin_amdgcn_wmma_f32_16x16x32_bf16(false, a, false, b1, (short)0, c1, false, false);
  }
#pragma unroll
  for (int r = 0; r < 8; ++r) {
    int m = m0 + r + half * 8;
    if (m >= M) continue;
    int na = n0 + l16, nb = n0 + 16 + l16;
    float v0 = c0[r] + bias[na];
    float v1 = c1[r] + bias[nb];
    if (GELU) { v0 = gelu_f(v0); v1 = gelu_f(v1); }
    size_t o0 = (size_t)m * N + na, o1 = (size_t)m * N + nb;
    if (RES) { v0 += res[o0]; v1 += res[o1]; }
    C[o0] = (OutT)v0; C[o1] = (OutT)v1;
  }
}

// ---------------- encoder attention: 8 waves = 8 heads, 16-query tile ----------------
// qkv is bf16 [M, 3E]; scores/softmax in fp32 LDS; output bf16 (feeds out-proj GEMM).
__global__ __launch_bounds__(256) void enc_attn_k(const __bf16* __restrict__ qkv,
                                                  const float* __restrict__ coords,
                                                  const float* __restrict__ gamma,
                                                  __bf16* __restrict__ out) {
  extern __shared__ float smem[];                 // 8 * 16 * Sc floats (256 KB of 320 KB WGP LDS)
  __shared__ float rsum[8][16];
  int b  = blockIdx.x / (Sc / 16);
  int qt = blockIdx.x % (Sc / 16);
  int lane = threadIdx.x & 31, head = threadIdx.x >> 5;
  int l16 = lane & 15, half = lane >> 4;
  int q0 = qt * 16;
  float* sc = smem + (size_t)head * 16 * Sc;

  // Q fragment (A-matrix 16x32, rows = queries, K-dim = head dim 32)
  const __bf16* Qb = qkv + ((size_t)(b * Sc + q0 + l16)) * (3 * Ec) + head * 32;
  int ka = half * 16;
  v8bf qlo = *(const v8bf*)(Qb + ka);
  v8bf qhi = *(const v8bf*)(Qb + ka + 8);
  v16bf aq = __builtin_shufflevector(qlo, qhi, 0, 1, 2, 3, 4, 5, 6, 7,
                                     8, 9, 10, 11, 12, 13, 14, 15);
  // per-row query coords for the dR bias
  float cqx[8], cqy[8];
#pragma unroll
  for (int r = 0; r < 8; ++r) {
    int m = q0 + r + half * 8;
    cqx[r] = coords[((size_t)b * Sc + m) * 2 + 0];
    cqy[r] = coords[((size_t)b * Sc + m) * 2 + 1];
  }
  float g = gamma[0];
  const float binv = 1.0f / (2.0f * SIG2v);

  for (int kt = 0; kt < Sc / 16; ++kt) {
    int k0 = kt * 16;
    const __bf16* Kb = qkv + ((size_t)(b * Sc + k0 + l16)) * (3 * Ec) + Ec + head * 32 + half * 16;
    v16bf bk = *(const v16bf*)Kb;
    v8f s = {};
    s = __builtin_amdgcn_wmma_f32_16x16x32_bf16(false, aq, false, bk, (short)0, s, false, false);
    float ckx = coords[((size_t)b * Sc + k0 + l16) * 2 + 0];
    float cky = coords[((size_t)b * Sc + k0 + l16) * 2 + 1];
#pragma unroll
    for (int r = 0; r < 8; ++r) {
      float dx = cqx[r] - ckx, dy = cqy[r] - cky;
      float biasv = -g * (dx * dx + dy * dy) * binv;
      sc[(size_t)(r + half * 8) * Sc + k0 + l16] = s[r] * HD_SCALE + biasv;
    }
  }
  __syncthreads();

  // two-pass softmax (unnormalized exp left in LDS; 1/sum folded into PV epilogue)
  for (int r = 0; r < 16; ++r) {
    float mx = -3.0e38f;
    for (int c = lane; c < Sc; c += 32) mx = fmaxf(mx, sc[(size_t)r * Sc + c]);
    mx = wave_max(mx);
    float sm = 0.f;
    for (int c = lane; c < Sc; c += 32) {
      float e = expf(sc[(size_t)r * Sc + c] - mx);
      sc[(size_t)r * Sc + c] = e;
      sm += e;
    }
    sm = wave_sum(sm);
    if (lane == 0) rsum[head][r] = sm;
  }
  __syncthreads();

  // PV: accumulate over keys in chunks of 32
  v8f o0 = {}, o1 = {};
  const __bf16* Vb = qkv + (size_t)b * Sc * (3 * Ec) + 2 * Ec + head * 32;
  for (int kc = 0; kc < Sc; kc += 32) {
    v16bf ap;
    int kb = kc + half * 16;
#pragma unroll
    for (int i = 0; i < 8; ++i) {
      ap[i]     = (__bf16)sc[(size_t)l16 * Sc + kb + i];
      ap[8 + i] = (__bf16)sc[(size_t)l16 * Sc + kb + 8 + i];
    }
    v16bf bv0, bv1;
#pragma unroll
    for (int i = 0; i < 16; ++i) {
      const __bf16* vr = Vb + (size_t)(kc + half * 16 + i) * (3 * Ec);
      bv0[i] = vr[l16];
      bv1[i] = vr[16 + l16];
    }
    o0 = __builtin_amdgcn_wmma_f32_16x16x32_bf16(false, ap, false, bv0, (short)0, o0, false, false);
    o1 = __builtin_amdgcn_wmma_f32_16x16x32_bf16(false, ap, false, bv1, (short)0, o1, false, false);
  }
#pragma unroll
  for (int r = 0; r < 8; ++r) {
    int m = r + half * 8;
    float inv = 1.0f / rsum[head][m];
    size_t row = ((size_t)(b * Sc + q0 + m)) * Ec + head * 32;
    out[row + l16]      = (__bf16)(o0[r] * inv);
    out[row + 16 + l16] = (__bf16)(o1[r] * inv);
  }
}

// ---------------- decoder attention (scalar, online softmax; tiny) ----------------
__global__ void dec_attn_k(const __bf16* __restrict__ Qb, int qs,
                           const __bf16* __restrict__ Kb, int ks,
                           const __bf16* __restrict__ Vb, int vs,
                           __bf16* __restrict__ out, int Tq, int Tk) {
  int idx = blockIdx.x * blockDim.x + threadIdx.x;
  if (idx >= Bc * Tq * Hh) return;
  int hh = idx % Hh;
  int q  = (idx / Hh) % Tq;
  int b  = idx / (Hh * Tq);
  const __bf16* qv = Qb + ((size_t)(b * Tq + q)) * qs + hh * 32;
  float qr[32], acc[32];
#pragma unroll
  for (int d = 0; d < 32; ++d) { qr[d] = (float)qv[d]; acc[d] = 0.f; }
  float mx = -3.0e38f, sum = 0.f;
  for (int k = 0; k < Tk; ++k) {
    const __bf16* kr = Kb + ((size_t)(b * Tk + k)) * ks + hh * 32;
    float s = 0.f;
#pragma unroll
    for (int d = 0; d < 32; ++d) s += qr[d] * (float)kr[d];
    s *= HD_SCALE;
    float nmx = fmaxf(mx, s);
    float corr = expf(mx - nmx);
    float p = expf(s - nmx);
    const __bf16* vr = Vb + ((size_t)(b * Tk + k)) * vs + hh * 32;
#pragma unroll
    for (int d = 0; d < 32; ++d) acc[d] = acc[d] * corr + p * (float)vr[d];
    sum = sum * corr + p;
    mx = nmx;
  }
  float inv = 1.0f / sum;
  __bf16* orow = out + ((size_t)(b * Tq + q)) * Ec + hh * 32;
#pragma unroll
  for (int d = 0; d < 32; ++d) orow[d] = (__bf16)(acc[d] * inv);
}

// ---------------- small heads (fp32 scalar; negligible work) ----------------
__global__ void parent_k(const float* __restrict__ mem, const float* __restrict__ pw,
                         const float* __restrict__ pb, float* __restrict__ out) {
  int row = blockIdx.x * blockDim.x + threadIdx.x;
  if (row >= Mrows) return;
  const float* r = mem + (size_t)row * Ec;
  float s = pb[0];
  for (int j = 0; j < Ec; ++j) s += r[j] * pw[j];
  out[row] = s;
}

__global__ void pool_k(const float* __restrict__ mem, float* __restrict__ g) {
  int idx = blockIdx.x * blockDim.x + threadIdx.x;
  if (idx >= Bc * Ec) return;
  int b = idx / Ec, e = idx % Ec;
  float s = 0.f;
  for (int t = 0; t < Sc; ++t) s += mem[((size_t)b * Sc + t) * Ec + e];
  g[idx] = s * (1.0f / Sc);
}

__global__ void cond_hid_k(const float* __restrict__ mem, const int* __restrict__ pidx,
                           const float* __restrict__ g, const float* __restrict__ c1w,
                           const float* __restrict__ c1b, float* __restrict__ hid) {
  int idx = blockIdx.x * blockDim.x + threadIdx.x;
  if (idx >= Bc * Ec) return;
  int b = idx / Ec, e = idx % Ec;
  const float* p = mem + ((size_t)b * Sc + pidx[b]) * Ec;
  const float* w = c1w + (size_t)e * (2 * Ec);
  float s = c1b[e];
  for (int j = 0; j < Ec; ++j) s += p[j] * w[j];
  for (int j = 0; j < Ec; ++j) s += g[b * Ec + j] * w[Ec + j];
  hid[idx] = gelu_f(s);
}

__global__ void cond_out_k(const float* __restrict__ hid, const float* __restrict__ c2w,
                           const float* __restrict__ c2b, float* __restrict__ cond) {
  int idx = blockIdx.x * blockDim.x + threadIdx.x;
  if (idx >= Bc * Ec) return;
  int b = idx / Ec, e = idx % Ec;
  const float* w = c2w + (size_t)e * Ec;
  float s = c2b[e];
  for (int j = 0; j < Ec; ++j) s += hid[b * Ec + j] * w[j];
  cond[idx] = s;
}

__global__ void tinit_k(const float* __restrict__ qe, const float* __restrict__ cond,
                        float* __restrict__ t) {
  int idx = blockIdx.x * blockDim.x + threadIdx.x;
  if (idx >= Bc * Kq * Ec) return;
  int e = idx % Ec, k = (idx / Ec) % Kq, b = idx / (Ec * Kq);
  t[idx] = qe[k * Ec + e] + cond[b * Ec + e];
}

__global__ void child_k(const float* __restrict__ t, const float* __restrict__ cw,
                        const float* __restrict__ cb, float* __restrict__ out) {
  int idx = blockIdx.x * blockDim.x + threadIdx.x;
  if (idx >= Bc * Kq * CDc) return;
  int cd = idx % CDc, row = idx / CDc;
  const float* r = t + (size_t)row * Ec;
  const float* w = cw + (size_t)cd * Ec;
  float s = cb[cd];
  for (int j = 0; j < Ec; ++j) s += r[j] * w[j];
  out[idx] = s;
}

__global__ void objhid_k(const float* __restrict__ t, const float* __restrict__ w1,
                         const float* __restrict__ b1, float* __restrict__ hid) {
  int idx = blockIdx.x * blockDim.x + threadIdx.x;
  if (idx >= Bc * Kq * Ec) return;
  int e = idx % Ec, row = idx / Ec;
  const float* r = t + (size_t)row * Ec;
  const float* w = w1 + (size_t)e * Ec;
  float s = b1[e];
  for (int j = 0; j < Ec; ++j) s += r[j] * w[j];
  hid[idx] = gelu_f(s);
}

__global__ void obj_k(const float* __restrict__ hid, const float* __restrict__ w2,
                      const float* __restrict__ b2, float* __restrict__ out) {
  int idx = blockIdx.x * blockDim.x + threadIdx.x;
  if (idx >= Bc * Kq) return;
  const float* r = hid + (size_t)idx * Ec;
  float s = b2[0];
  for (int j = 0; j < Ec; ++j) s += r[j] * w2[j];
  out[idx] = s;
}

// ---------------- host orchestration ----------------
extern "C" void kernel_launch(void* const* d_in, const int* in_sizes, int n_in,
                              void* d_out, int out_size, void* d_ws, size_t ws_size,
                              hipStream_t stream) {
  (void)in_sizes; (void)n_in; (void)out_size; (void)ws_size;
  const float* x          = (const float*)d_in[0];
  const int*   parent_idx = (const int*)d_in[2];
  const float* in_w = (const float*)d_in[3];
  const float* in_b = (const float*)d_in[4];
  const float* bn_w = (const float*)d_in[5];
  const float* bn_b = (const float*)d_in[6];
  const float* dr_gamma = (const float*)d_in[7];
  const float* e_qkv_w = (const float*)d_in[8];
  const float* e_qkv_b = (const float*)d_in[9];
  const float* e_out_w = (const float*)d_in[10];
  const float* e_out_b = (const float*)d_in[11];
  const float* e_l1_w = (const float*)d_in[12];
  const float* e_l1_b = (const float*)d_in[13];
  const float* e_l2_w = (const float*)d_in[14];
  const float* e_l2_b = (const float*)d_in[15];
  const float* e_n1_w = (const float*)d_in[16];
  const float* e_n1_b = (const float*)d_in[17];
  const float* e_n2_w = (const float*)d_in[18];
  const float* e_n2_b = (const float*)d_in[19];
  const float* parent_w = (const float*)d_in[20];
  const float* parent_b = (const float*)d_in[21];
  const float* query_embed = (const float*)d_in[22];
  const float* c1_w = (const float*)d_in[23];
  const float* c1_b = (const float*)d_in[24];
  const float* c2_w = (const float*)d_in[25];
  const float* c2_b = (const float*)d_in[26];
  const float* d_sqkv_w = (const float*)d_in[27];
  const float* d_sqkv_b = (const float*)d_in[28];
  const float* d_sout_w = (const float*)d_in[29];
  const float* d_sout_b = (const float*)d_in[30];
  const float* d_cqkv_w = (const float*)d_in[31];
  const float* d_cqkv_b = (const float*)d_in[32];
  const float* d_cout_w = (const float*)d_in[33];
  const float* d_cout_b = (const float*)d_in[34];
  const float* d_l1_w = (const float*)d_in[35];
  const float* d_l1_b = (const float*)d_in[36];
  const float* d_l2_w = (const float*)d_in[37];
  const float* d_l2_b = (const float*)d_in[38];
  const float* d_n1_w = (const float*)d_in[39];
  const float* d_n1_b = (const float*)d_in[40];
  const float* d_n2_w = (const float*)d_in[41];
  const float* d_n2_b = (const float*)d_in[42];
  const float* d_n3_w = (const float*)d_in[43];
  const float* d_n3_b = (const float*)d_in[44];
  const float* child_w = (const float*)d_in[45];
  const float* child_b = (const float*)d_in[46];
  const float* o1_w = (const float*)d_in[47];
  const float* o1_b = (const float*)d_in[48];
  const float* o2_w = (const float*)d_in[49];
  const float* o2_b = (const float*)d_in[50];
  float* out = (float*)d_out;

  // ---- workspace carve-up (64B aligned) ----
  char* base = (char*)d_ws;
  size_t off = 0;
  auto allocf = [&](size_t n) {
    off = (off + 63) & ~(size_t)63; float* p = (float*)(base + off); off += n * 4; return p;
  };
  auto allocb = [&](size_t n) {
    off = (off + 63) & ~(size_t)63; __bf16* p = (__bf16*)(base + off); off += n * 2; return p;
  };
  // fp32 master buffers
  float* h      = allocf((size_t)Mrows * Ec);
  float* coords = allocf((size_t)Mrows * 2);
  float* t      = allocf((size_t)Bc * Kq * Ec);
  float* gbuf   = allocf((size_t)Bc * Ec);
  float* hid    = allocf((size_t)Bc * Ec);
  float* cond   = allocf((size_t)Bc * Ec);
  float* ohid   = allocf((size_t)Bc * Kq * Ec);
  // bf16 activation buffers
  __bf16* hn_bf    = allocb((size_t)Mrows * Ec);
  __bf16* qkv_bf   = allocb((size_t)Mrows * 3 * Ec);
  __bf16* attn_bf  = allocb((size_t)Mrows * Ec);
  __bf16* ff_bf    = allocb((size_t)Mrows * Fc);
  __bf16* hbf      = allocb((size_t)Mrows * Ec);      // bf16 copy of memory
  __bf16* tn_bf    = allocb((size_t)Bc * Kq * Ec);
  __bf16* dqkv_bf  = allocb((size_t)Bc * Kq * 3 * Ec);
  __bf16* dattn_bf = allocb((size_t)Bc * Kq * Ec);
  __bf16* dq_bf    = allocb((size_t)Bc * Kq * Ec);
  __bf16* kv_bf    = allocb((size_t)Mrows * 2 * Ec);
  __bf16* dff_bf   = allocb((size_t)Bc * Kq * Fc);
  // bf16 weights (converted once per launch)
  __bf16* wqkv_e = allocb((size_t)LEc * 3 * Ec * Ec);
  __bf16* wout_e = allocb((size_t)LEc * Ec * Ec);
  __bf16* wl1_e  = allocb((size_t)LEc * Fc * Ec);
  __bf16* wl2_e  = allocb((size_t)LEc * Ec * Fc);
  __bf16* wsqkv_d = allocb((size_t)LDc * 3 * Ec * Ec);
  __bf16* wsout_d = allocb((size_t)LDc * Ec * Ec);
  __bf16* wcqkv_d = allocb((size_t)LDc * 3 * Ec * Ec);
  __bf16* wcout_d = allocb((size_t)LDc * Ec * Ec);
  __bf16* wl1_d   = allocb((size_t)LDc * Fc * Ec);
  __bf16* wl2_d   = allocb((size_t)LDc * Ec * Fc);

  auto cvt = [&](const float* src, __bf16* dst, int n) {
    cvt_bf16_k<<<(n + 255) / 256, 256, 0, stream>>>(src, dst, n);
  };
  cvt(e_qkv_w, wqkv_e, LEc * 3 * Ec * Ec);
  cvt(e_out_w, wout_e, LEc * Ec * Ec);
  cvt(e_l1_w, wl1_e, LEc * Fc * Ec);
  cvt(e_l2_w, wl2_e, LEc * Ec * Fc);
  cvt(d_sqkv_w, wsqkv_d, LDc * 3 * Ec * Ec);
  cvt(d_sout_w, wsout_d, LDc * Ec * Ec);
  cvt(d_cqkv_w, wcqkv_d, LDc * 3 * Ec * Ec);
  cvt(d_cout_w, wcout_d, LDc * Ec * Ec);
  cvt(d_l1_w, wl1_d, LDc * Fc * Ec);
  cvt(d_l2_w, wl2_d, LDc * Ec * Fc);

  // input projection
  input_proj_k<<<(Mrows * Ec) / 256, 256, 0, stream>>>(x, in_w, in_b, bn_w, bn_b, h, coords);

  // encoder
  const size_t attn_lds = (size_t)8 * 16 * Sc * sizeof(float);  // 256 KB of 320 KB WGP LDS
  for (int l = 0; l < LEc; ++l) {
    layernorm_k<<<Mrows / 8, 256, 0, stream>>>(h, hn_bf, e_n1_w + l * Ec, e_n1_b + l * Ec, Mrows);
    gemm_k<false, false, __bf16><<<dim3(3 * Ec / 64, Mrows / 64), 256, 0, stream>>>(
        hn_bf, wqkv_e + (size_t)l * 3 * Ec * Ec, e_qkv_b + l * 3 * Ec, nullptr, qkv_bf,
        Mrows, 3 * Ec, Ec);
    enc_attn_k<<<Bc * (Sc / 16), 256, attn_lds, stream>>>(qkv_bf, coords, dr_gamma, attn_bf);
    gemm_k<false, true, float><<<dim3(Ec / 64, Mrows / 64), 256, 0, stream>>>(
        attn_bf, wout_e + (size_t)l * Ec * Ec, e_out_b + l * Ec, h, h, Mrows, Ec, Ec);
    layernorm_k<<<Mrows / 8, 256, 0, stream>>>(h, hn_bf, e_n2_w + l * Ec, e_n2_b + l * Ec, Mrows);
    gemm_k<true, false, __bf16><<<dim3(Fc / 64, Mrows / 64), 256, 0, stream>>>(
        hn_bf, wl1_e + (size_t)l * Fc * Ec, e_l1_b + l * Fc, nullptr, ff_bf, Mrows, Fc, Ec);
    gemm_k<false, true, float><<<dim3(Ec / 64, Mrows / 64), 256, 0, stream>>>(
        ff_bf, wl2_e + (size_t)l * Ec * Fc, e_l2_b + l * Ec, h, h, Mrows, Ec, Fc);
  }
  // h is now `memory`; keep a bf16 copy for cross-attention K/V projections
  cvt(h, hbf, Mrows * Ec);

  // parent logits -> out[0 .. B*S)
  parent_k<<<Mrows / 256, 256, 0, stream>>>(h, parent_w, parent_b, out);

  // conditional decoder query construction
  pool_k<<<(Bc * Ec) / 256, 256, 0, stream>>>(h, gbuf);
  cond_hid_k<<<(Bc * Ec) / 256, 256, 0, stream>>>(h, parent_idx, gbuf, c1_w, c1_b, hid);
  cond_out_k<<<(Bc * Ec) / 256, 256, 0, stream>>>(hid, c2_w, c2_b, cond);
  tinit_k<<<(Bc * Kq * Ec) / 256, 256, 0, stream>>>(query_embed, cond, t);

  // decoder
  const int Mt = Bc * Kq;  // 128
  for (int l = 0; l < LDc; ++l) {
    // self-attention
    layernorm_k<<<Mt / 8, 256, 0, stream>>>(t, tn_bf, d_n1_w + l * Ec, d_n1_b + l * Ec, Mt);
    gemm_k<false, false, __bf16><<<dim3(3 * Ec / 64, (Mt + 63) / 64), 256, 0, stream>>>(
        tn_bf, wsqkv_d + (size_t)l * 3 * Ec * Ec, d_sqkv_b + l * 3 * Ec, nullptr, dqkv_bf,
        Mt, 3 * Ec, Ec);
    dec_attn_k<<<(Bc * Kq * Hh + 255) / 256, 256, 0, stream>>>(
        dqkv_bf, 3 * Ec, dqkv_bf + Ec, 3 * Ec, dqkv_bf + 2 * Ec, 3 * Ec, dattn_bf, Kq, Kq);
    gemm_k<false, true, float><<<dim3(Ec / 64, (Mt + 63) / 64), 256, 0, stream>>>(
        dattn_bf, wsout_d + (size_t)l * Ec * Ec, d_sout_b + l * Ec, t, t, Mt, Ec, Ec);
    // cross-attention: q from tn, k/v projected from memory
    layernorm_k<<<Mt / 8, 256, 0, stream>>>(t, tn_bf, d_n2_w + l * Ec, d_n2_b + l * Ec, Mt);
    gemm_k<false, false, __bf16><<<dim3(Ec / 64, (Mt + 63) / 64), 256, 0, stream>>>(
        tn_bf, wcqkv_d + (size_t)l * 3 * Ec * Ec, d_cqkv_b + l * 3 * Ec, nullptr, dq_bf,
        Mt, Ec, Ec);
    gemm_k<false, false, __bf16><<<dim3(2 * Ec / 64, Mrows / 64), 256, 0, stream>>>(
        hbf, wcqkv_d + (size_t)l * 3 * Ec * Ec + (size_t)Ec * Ec,
        d_cqkv_b + l * 3 * Ec + Ec, nullptr, kv_bf, Mrows, 2 * Ec, Ec);
    dec_attn_k<<<(Bc * Kq * Hh + 255) / 256, 256, 0, stream>>>(
        dq_bf, Ec, kv_bf, 2 * Ec, kv_bf + Ec, 2 * Ec, dattn_bf, Kq, Sc);
    gemm_k<false, true, float><<<dim3(Ec / 64, (Mt + 63) / 64), 256, 0, stream>>>(
        dattn_bf, wcout_d + (size_t)l * Ec * Ec, d_cout_b + l * Ec, t, t, Mt, Ec, Ec);
    // FFN
    layernorm_k<<<Mt / 8, 256, 0, stream>>>(t, tn_bf, d_n3_w + l * Ec, d_n3_b + l * Ec, Mt);
    gemm_k<true, false, __bf16><<<dim3(Fc / 64, (Mt + 63) / 64), 256, 0, stream>>>(
        tn_bf, wl1_d + (size_t)l * Fc * Ec, d_l1_b + l * Fc, nullptr, dff_bf, Mt, Fc, Ec);
    gemm_k<false, true, float><<<dim3(Ec / 64, (Mt + 63) / 64), 256, 0, stream>>>(
        dff_bf, wl2_d + (size_t)l * Ec * Fc, d_l2_b + l * Ec, t, t, Mt, Ec, Fc);
  }

  // heads: child_feat -> out[B*S .. B*S + B*K*CD), obj -> after that
  child_k<<<(Bc * Kq * CDc + 255) / 256, 256, 0, stream>>>(t, child_w, child_b, out + Mrows);
  objhid_k<<<(Bc * Kq * Ec) / 256, 256, 0, stream>>>(t, o1_w, o1_b, ohid);
  obj_k<<<(Bc * Kq + 255) / 256, 256, 0, stream>>>(ohid, o2_w, o2_b,
                                                   out + Mrows + Bc * Kq * CDc);
}
